// NetworkWrapper_52501680226347
// MI455X (gfx1250) — matrix operand order repaired and checked by reading, not compiled, not run on hardware
//
#include <hip/hip_runtime.h>
#include <hip/hip_bf16.h>

// ---------------------------------------------------------------------------
// Problem constants (from the reference)
// ---------------------------------------------------------------------------
#define T_NUM 65536      // target rows
#define S_NUM 2560       // support rows
#define FEAT  128        // feature dim (K)
#define NCLS  40         // classes (= class_num_per_batch)
#define GRP   64         // supports per class
#define ROW_TILES (T_NUM / 16)   // 4096

typedef __attribute__((ext_vector_type(16))) _Float16 v16h;
typedef __attribute__((ext_vector_type(8)))  _Float16 h8;
typedef __attribute__((ext_vector_type(8)))  float    v8f;

__device__ __forceinline__ h8 ld8(const _Float16* p) {
    return *reinterpret_cast<const h8*>(p);
}
__device__ __forceinline__ v16h cat8(h8 a, h8 b) {
    v16h r;
#pragma unroll
    for (int i = 0; i < 8; ++i) { r[i] = a[i]; r[i + 8] = b[i]; }
    return r;
}

// ---------------------------------------------------------------------------
// K1: normalize support features, split f32 -> f16 hi + f16 lo residual
//     one block (128 threads) per row
// ---------------------------------------------------------------------------
__global__ void prep_support_kernel(const float* __restrict__ in,
                                    _Float16* __restrict__ hi,
                                    _Float16* __restrict__ lo) {
    const int row = blockIdx.x;
    const int k   = threadIdx.x;
    float x = in[(size_t)row * FEAT + k];
    float s = x * x;
#pragma unroll
    for (int m = 16; m >= 1; m >>= 1) s += __shfl_xor(s, m, 32);
    __shared__ float part[4];
    if ((threadIdx.x & 31) == 0) part[threadIdx.x >> 5] = s;
    __syncthreads();
    float tot = part[0] + part[1] + part[2] + part[3];
    float xn  = x * rsqrtf(tot + 1e-12f);
    _Float16 h = (_Float16)xn;
    hi[(size_t)row * FEAT + k] = h;
    lo[(size_t)row * FEAT + k] = (_Float16)(xn - (float)h);
}

// ---------------------------------------------------------------------------
// K2: normalize target features, split hi/lo, fused direct linear classifier
//     logits[row][c] = dot(tf_norm[row], W[c]) + b[c]
// ---------------------------------------------------------------------------
__global__ void prep_target_kernel(const float* __restrict__ in,
                                   _Float16* __restrict__ hi,
                                   _Float16* __restrict__ lo,
                                   const float* __restrict__ W,
                                   const float* __restrict__ b,
                                   float* __restrict__ logits) {
    const int row = blockIdx.x;
    const int k   = threadIdx.x;
    __shared__ float part[4];
    __shared__ float xsh[FEAT];
    float x = in[(size_t)row * FEAT + k];
    float s = x * x;
#pragma unroll
    for (int m = 16; m >= 1; m >>= 1) s += __shfl_xor(s, m, 32);
    if ((threadIdx.x & 31) == 0) part[threadIdx.x >> 5] = s;
    __syncthreads();
    float tot = part[0] + part[1] + part[2] + part[3];
    float xn  = x * rsqrtf(tot + 1e-12f);
    _Float16 h = (_Float16)xn;
    hi[(size_t)row * FEAT + k] = h;
    lo[(size_t)row * FEAT + k] = (_Float16)(xn - (float)h);
    xsh[k] = xn;
    __syncthreads();
    if (k < NCLS) {
        const float4* w = reinterpret_cast<const float4*>(W + (size_t)k * FEAT);
        float acc = b[k];
#pragma unroll
        for (int i = 0; i < FEAT / 4; ++i) {
            float4 wv = w[i];
            acc += xsh[4 * i + 0] * wv.x + xsh[4 * i + 1] * wv.y +
                   xsh[4 * i + 2] * wv.z + xsh[4 * i + 3] * wv.w;
        }
        logits[(size_t)row * NCLS + k] = acc;
    }
}

// ---------------------------------------------------------------------------
// K3: cosine-similarity GEMM via V_WMMA_F32_16X16X32_F16 with split-f16
//     (hi*hi + lo*hi + hi*lo) ~ f32 accuracy at f16 matrix throughput.
//     One wave -> 16 rows x 64 cols (= one class group). All global accesses
//     use ONE base pointer per stream + compile-time immediate offsets so the
//     backend emits base+offset:imm loads/stores instead of 64-bit adds.
// ---------------------------------------------------------------------------
__global__ void __launch_bounds__(256)
cosine_gemm_kernel(const _Float16* __restrict__ tf_hi,
                   const _Float16* __restrict__ tf_lo,
                   const _Float16* __restrict__ sf_hi,
                   const _Float16* __restrict__ sf_lo,
                   float* __restrict__ out_ip,
                   float* __restrict__ grouped) {
    const int lane   = threadIdx.x & 31;
    const int wave   = threadIdx.x >> 5;
    const int waveId = blockIdx.x * 8 + wave;
    const int rowTile = waveId / NCLS;            // 0..4095
    const int cls     = waveId - rowTile * NCLS;  // 0..39
    const int g   = lane >> 4;                    // half-wave group
    const int l16 = lane & 15;

    // ---- stream base pointers (computed once; all further offsets are imm) --
    // A (16x32 f16): lane group 0 holds K {c*32+0..7, c*32+16..23},
    //                group 1 holds {c*32+8..15, c*32+24..31}
    const size_t arow = (size_t)(rowTile * 16 + l16) * FEAT + (size_t)(g * 8);
    const _Float16* abase_hi = tf_hi + arow;
    const _Float16* abase_lo = tf_lo + arow;
    // B (32x16): lane = N = l16; group g holds K {c*32+g*8..+7, c*32+16+g*8..+7}
    const size_t brow = (size_t)(cls * GRP + l16) * FEAT + (size_t)(g * 8);
    const _Float16* bbase_hi = sf_hi + brow;
    const _Float16* bbase_lo = sf_lo + brow;

    v16h Ahi[4], Alo[4];
#pragma unroll
    for (int c = 0; c < 4; ++c) {
        Ahi[c] = cat8(ld8(abase_hi + 32 * c), ld8(abase_hi + 32 * c + 16));
        Alo[c] = cat8(ld8(abase_lo + 32 * c), ld8(abase_lo + 32 * c + 16));
    }

    const int row0 = rowTile * 16 + g * 8;        // first row this half-wave owns
    float* obase = out_ip + (size_t)row0 * S_NUM + (size_t)(cls * GRP + l16);

    v8f sum = {};
#pragma unroll
    for (int n = 0; n < 4; ++n) {
        v8f acc = {};
#pragma unroll
        for (int c = 0; c < 4; ++c) {
            const int off = n * 16 * FEAT + 32 * c;   // compile-time constant
            v16h Bh = cat8(ld8(bbase_hi + off), ld8(bbase_hi + off + 16));
            v16h Bl = cat8(ld8(bbase_lo + off), ld8(bbase_lo + off + 16));
            acc = __builtin_amdgcn_wmma_f32_16x16x32_f16(false, Ahi[c], false, Bh,
                                                         (short)0, acc, false, false);
            acc = __builtin_amdgcn_wmma_f32_16x16x32_f16(false, Alo[c], false, Bh,
                                                         (short)0, acc, false, false);
            acc = __builtin_amdgcn_wmma_f32_16x16x32_f16(false, Ahi[c], false, Bl,
                                                         (short)0, acc, false, false);
        }
        // C/D layout: VGPR v, lanes 0-15 -> M=v, lanes 16-31 -> M=8+v; N = l16
#pragma unroll
        for (int v = 0; v < 8; ++v)
            obase[(size_t)(v * S_NUM + n * 16)] = acc[v];   // imm offsets
        sum += acc;
    }

    // group mean: reduce 16 lanes of each half-wave (xor masks stay in-group)
#pragma unroll
    for (int m = 8; m >= 1; m >>= 1) {
#pragma unroll
        for (int v = 0; v < 8; ++v) sum[v] += __shfl_xor(sum[v], m, 32);
    }
    if (l16 == 0) {
        float* gbase = grouped + (size_t)row0 * NCLS + cls;
#pragma unroll
        for (int v = 0; v < 8; ++v)
            gbase[(size_t)(v * NCLS)] = sum[v] * (1.0f / GRP);
    }
}

// ---------------------------------------------------------------------------
// K4: argmax over 40 class means (first-max tie-break, like jnp.argmax),
//     then map through class labels; store prediction as float
// ---------------------------------------------------------------------------
__global__ void preds_kernel(const float* __restrict__ grouped,
                             const int* __restrict__ labels,
                             float* __restrict__ out_preds) {
    const int t = blockIdx.x * blockDim.x + threadIdx.x;
    if (t >= T_NUM) return;
    const float* gp = grouped + (size_t)t * NCLS;
    float bv = gp[0];
    int   bi = 0;
#pragma unroll
    for (int c = 1; c < NCLS; ++c) {
        float v = gp[c];
        if (v > bv) { bv = v; bi = c; }
    }
    out_preds[t] = (float)labels[bi * GRP];
}

// ---------------------------------------------------------------------------
// launcher
// ---------------------------------------------------------------------------
extern "C" void kernel_launch(void* const* d_in, const int* in_sizes, int n_in,
                              void* d_out, int out_size, void* d_ws, size_t ws_size,
                              hipStream_t stream) {
    const float* tf     = (const float*)d_in[0];
    const float* sf     = (const float*)d_in[1];
    const float* W      = (const float*)d_in[2];
    const float* b      = (const float*)d_in[3];
    const int*   labels = (const int*)d_in[4];

    // outputs, concatenated: inner_product [T,S], preds [T], logits [T,NCLS]
    float* out_ip     = (float*)d_out;
    float* out_preds  = out_ip + (size_t)T_NUM * S_NUM;
    float* out_logits = out_preds + T_NUM;

    // workspace layout (256B aligned slices)
    char* ws = (char*)d_ws;
    size_t off = 0;
    auto take = [&](size_t bytes) {
        void* p = ws + off;
        off += (bytes + 255) & ~(size_t)255;
        return p;
    };
    _Float16* tf_hi = (_Float16*)take((size_t)T_NUM * FEAT * sizeof(_Float16));
    _Float16* tf_lo = (_Float16*)take((size_t)T_NUM * FEAT * sizeof(_Float16));
    _Float16* sf_hi = (_Float16*)take((size_t)S_NUM * FEAT * sizeof(_Float16));
    _Float16* sf_lo = (_Float16*)take((size_t)S_NUM * FEAT * sizeof(_Float16));
    float*    grouped = (float*)take((size_t)T_NUM * NCLS * sizeof(float));
    (void)in_sizes; (void)n_in; (void)out_size; (void)ws_size;

    prep_support_kernel<<<S_NUM, FEAT, 0, stream>>>(sf, sf_hi, sf_lo);
    prep_target_kernel<<<T_NUM, FEAT, 0, stream>>>(tf, tf_hi, tf_lo, W, b, out_logits);

    const int totalWaves = ROW_TILES * NCLS;   // 163840 waves, 8 per block
    cosine_gemm_kernel<<<totalWaves / 8, 256, 0, stream>>>(tf_hi, tf_lo, sf_hi, sf_lo,
                                                           out_ip, grouped);
    preds_kernel<<<(T_NUM + 255) / 256, 256, 0, stream>>>(grouped, labels, out_preds);
}
